// MemoryReader_56581899157854
// MI455X (gfx1250) — matrix operand order repaired and checked by reading, not compiled, not run on hardware
//
#include <hip/hip_runtime.h>
#include <hip/hip_bf16.h>

typedef __attribute__((ext_vector_type(16))) _Float16 v16h;
typedef __attribute__((ext_vector_type(8)))  _Float16 v8h;
typedef __attribute__((ext_vector_type(8)))  float    v8f;

#define TAU_F 0.07f
#define CKDIM 128
#define TOPK  32
#define WAVES 4
#define MTILE 32   // queries per workgroup (2 x 16-row WMMA tiles)

// ---------------------------------------------------------------------------
// Kernel 1: L2-normalize along channel dim and transpose to position-major f16
// X: [B][C][N] fp32  ->  Y: [B*N][C] f16   (norm denom = max(||x||, 1e-12))
// ---------------------------------------------------------------------------
__global__ void mr_norm_t_kernel(const float* __restrict__ X,
                                 _Float16* __restrict__ Y,
                                 int N, int BN) {
    int t = blockIdx.x * blockDim.x + threadIdx.x;
    if (t >= BN) return;
    int b = t / N;
    int n = t - b * N;
    const float* x = X + (size_t)b * CKDIM * N + n;
    float s = 0.f;
    #pragma unroll 8
    for (int c = 0; c < CKDIM; ++c) {
        float v = x[(size_t)c * N];
        s += v * v;
    }
    float scale = 1.0f / fmaxf(sqrtf(s), 1e-12f);
    _Float16* y = Y + (size_t)t * CKDIM;
    #pragma unroll 8
    for (int c = 0; c < CKDIM; ++c)
        y[c] = (_Float16)(x[(size_t)c * N] * scale);
}

// ---------------------------------------------------------------------------
// Kernel 2: per 32-query tile: Qn^T Kn via v_wmma_f32_16x16x32_f16 (8 / tile),
// register ping-pong (compile-time indexed -> no scratch spills),
// streaming top-32, merge + softmax.
// ---------------------------------------------------------------------------
__global__ __launch_bounds__(128)
void mr_attn_topk_kernel(const _Float16* __restrict__ qn,
                         const _Float16* __restrict__ kn,
                         int Nq, int Nm,
                         int* __restrict__ out_idx,
                         float* __restrict__ out_w) {
    const int b     = blockIdx.y;
    const int qbase = blockIdx.x * MTILE;
    const _Float16* Aq = qn + ((size_t)b * Nq + qbase) * CKDIM;
    const _Float16* Bk = kn + (size_t)b * Nm * CKDIM;

    const int tid  = threadIdx.x;
    const int wave = tid >> 5;   // 0..3 (wave32)
    const int lane = tid & 31;

    __shared__ float sc[WAVES][MTILE][16];      //  8 KB
    __shared__ float topv[WAVES][MTILE][TOPK];  // 16 KB
    __shared__ int   topi[WAVES][MTILE][TOPK];  // 16 KB
    __shared__ float fin_v[MTILE][TOPK];        //  4 KB
    __shared__ int   fin_i[MTILE][TOPK];        //  4 KB

    for (int i = tid; i < WAVES * MTILE * TOPK; i += 128) {
        ((float*)topv)[i] = -1e30f;
        ((int*)topi)[i]   = 0;
    }
    __syncthreads();

    // ---- A fragments (resident whole kernel): two 16x32 f16 tiles per k-chunk.
    // A layout (matches B): lane<16 -> K 0..15; lane>=16 -> K 16..31 (contiguous per lane)
    const int arow = lane & 15;
    const int akb  = (lane < 16) ? 0 : 16;
    v16h afrag[2][4];
    #pragma unroll
    for (int m = 0; m < 2; ++m) {
        const _Float16* aptr = Aq + (m * 16 + arow) * CKDIM;
        #pragma unroll
        for (int kk = 0; kk < 4; ++kk) {
            v8h lo = *(const v8h*)(aptr + kk * 32 + akb);
            v8h hi = *(const v8h*)(aptr + kk * 32 + akb + 8);
            afrag[m][kk] = __builtin_shufflevector(lo, hi, 0, 1, 2, 3, 4, 5, 6, 7,
                                                   8, 9, 10, 11, 12, 13, 14, 15);
        }
    }

    // ---- B fragment addressing: 32x16 f16, N = lane&15,
    // lanes 0..15 hold K 0..15, lanes 16..31 hold K 16..31 (contiguous per lane).
    const int bcol = lane & 15;
    const int bkb  = (lane < 16) ? 0 : 16;

    // per-thread top-32 list owner state: slot s = tid>>5, row r = tid&31
    float curMin = -1e30f;
    int   curMinPos = 0;
    const int os   = tid >> 5;
    const int orow = tid & 31;

    const int nIter = Nm / (16 * WAVES);   // 320 (even)

    // load a 32x16 B tile (4 k-chunks) into named registers + prefetch ahead
    auto loadTile = [&](int nt, v16h (&bf)[4]) __attribute__((always_inline)) {
        const _Float16* brow = Bk + (size_t)(nt * 16 + bcol) * CKDIM;
        __builtin_prefetch(brow + (size_t)WAVES * 16 * CKDIM, 0, 1);
        #pragma unroll
        for (int kk = 0; kk < 4; ++kk)
            bf[kk] = *(const v16h*)(brow + kk * 32 + bkb);
    };

    // consume one tile: 8 WMMAs, score store, barrier, top-k update, barrier
    auto process = [&](int it, v16h (&bf)[4]) __attribute__((always_inline)) {
        v8f acc0 = {}, acc1 = {};
        #pragma unroll
        for (int kk = 0; kk < 4; ++kk) {
            acc0 = __builtin_amdgcn_wmma_f32_16x16x32_f16(
                       false, afrag[0][kk], false, bf[kk],
                       (short)0, acc0, false, false);
            acc1 = __builtin_amdgcn_wmma_f32_16x16x32_f16(
                       false, afrag[1][kk], false, bf[kk],
                       (short)0, acc1, false, false);
        }
        // C/D layout: VGPR r holds row (r + 8*(lane>>4)), col = lane&15
        const int rbase = (lane >> 4) * 8;
        #pragma unroll
        for (int r = 0; r < 8; ++r) {
            sc[wave][rbase + r][bcol]      = acc0[r];
            sc[wave][16 + rbase + r][bcol] = acc1[r];
        }
        __syncthreads();

        // top-k update: all 128 threads own one (slot, row) list
        const int mb = (it * WAVES + os) * 16;
        #pragma unroll 4
        for (int c = 0; c < 16; ++c) {
            float v = sc[os][orow][c];
            if (v > curMin) {
                topv[os][orow][curMinPos] = v;
                topi[os][orow][curMinPos] = mb + c;
                float mn = topv[os][orow][0]; int mp = 0;
                for (int j = 1; j < TOPK; ++j) {
                    float tv = topv[os][orow][j];
                    if (tv < mn) { mn = tv; mp = j; }
                }
                curMin = mn; curMinPos = mp;
            }
        }
        __syncthreads();
    };

    // ---- main loop, unrolled x2 with NAMED ping-pong buffers so every
    // fragment index is a compile-time constant (stays in VGPRs, no spills)
    v16h b0[4], b1[4];
    loadTile(wave, b0);                                  // tile for it = 0

    for (int it = 0; it < nIter; it += 2) {
        loadTile(((it + 1) * WAVES) + wave, b1);         // it+1 always < nIter
        process(it, b0);
        if (it + 2 < nIter)
            loadTile(((it + 2) * WAVES) + wave, b0);
        process(it + 1, b1);
    }

    // ---- merge 4 partial lists per row -> final top-32, then softmax(score/tau)
    if (tid < MTILE) {
        const int r = tid;
        for (int k = 0; k < TOPK; ++k) {
            float best = -1e30f; int bs = 0, bj = 0;
            for (int s = 0; s < WAVES; ++s)
                for (int j = 0; j < TOPK; ++j) {
                    float v = topv[s][r][j];
                    if (v > best) { best = v; bs = s; bj = j; }
                }
            fin_v[r][k] = best;
            fin_i[r][k] = topi[bs][r][bj];
            topv[bs][r][bj] = -1e31f;
        }
        const float mx = fin_v[r][0];
        float sum = 0.f;
        for (int k = 0; k < TOPK; ++k) {
            float e = __expf((fin_v[r][k] - mx) * (1.0f / TAU_F));
            fin_v[r][k] = e;
            sum += e;
        }
        const float inv = 1.0f / sum;
        const size_t o = ((size_t)b * Nq + qbase + r) * TOPK;
        for (int k = 0; k < TOPK; ++k) {
            out_w[o + k]   = fin_v[r][k] * inv;
            out_idx[o + k] = fin_i[r][k];
        }
    }
}

// ---------------------------------------------------------------------------
// Kernel 3: out[b][c][q] = sum_k w[b][q][k] * V[b][c][idx[b][q][k]]
// V is L2-resident (42 MB < 192 MB)
// ---------------------------------------------------------------------------
__global__ void mr_read_kernel(const float* __restrict__ V,
                               const int* __restrict__ idx,
                               const float* __restrict__ w,
                               float* __restrict__ out,
                               int Nq, int Nm, int Cv, long total) {
    long t = (long)blockIdx.x * blockDim.x + threadIdx.x;
    if (t >= total) return;
    int q = (int)(t % Nq);
    long bc = t / Nq;
    int c = (int)(bc % Cv);
    int b = (int)(bc / Cv);
    const float* vrow = V + ((size_t)b * Cv + c) * Nm;
    const size_t o = ((size_t)b * Nq + q) * TOPK;
    float acc = 0.f;
    #pragma unroll
    for (int k = 0; k < TOPK; ++k)
        acc += w[o + k] * vrow[idx[o + k]];
    out[t] = acc;
}

// ---------------------------------------------------------------------------
extern "C" void kernel_launch(void* const* d_in, const int* in_sizes, int n_in,
                              void* d_out, int out_size, void* d_ws, size_t ws_size,
                              hipStream_t stream) {
    const float* Q = (const float*)d_in[0];   // [2,128,64,64]
    const float* K = (const float*)d_in[1];   // [2,128,64,320]
    const float* V = (const float*)d_in[2];   // [2,257,64,320]
    float* out = (float*)d_out;               // [2,257,64,64]

    const int B = 2, Nq = 64 * 64, Nm = 64 * 320, Cv = 257;

    // workspace layout (14 MB total)
    char* ws = (char*)d_ws;
    _Float16* qn  = (_Float16*)ws;                          //  2 MB: [B*Nq][128] f16
    _Float16* kn  = (_Float16*)(ws + ((size_t)2  << 20));   // 10 MB: [B*Nm][128] f16
    int*      idx = (int*)     (ws + ((size_t)12 << 20));   //  1 MB: [B][Nq][32]
    float*    wgt = (float*)   (ws + ((size_t)13 << 20));   //  1 MB: [B][Nq][32]

    {
        int bn = B * Nq;
        mr_norm_t_kernel<<<(bn + 255) / 256, 256, 0, stream>>>(Q, qn, Nq, bn);
    }
    {
        int bn = B * Nm;
        mr_norm_t_kernel<<<(bn + 255) / 256, 256, 0, stream>>>(K, kn, Nm, bn);
    }
    {
        dim3 grid(Nq / MTILE, B);   // 128 x 2 workgroups, 128 threads (4 wave32)
        mr_attn_topk_kernel<<<grid, 128, 0, stream>>>(qn, kn, Nq, Nm, idx, wgt);
    }
    {
        long total = (long)B * Cv * Nq;
        mr_read_kernel<<<(int)((total + 255) / 256), 256, 0, stream>>>(
            V, idx, wgt, out, Nq, Nm, Cv, total);
    }
}
